// StructuredAttention_65180423684702
// MI455X (gfx1250) — matrix operand order repaired and checked by reading, not compiled, not run on hardware
//
#include <hip/hip_runtime.h>
#include <math.h>

// Problem constants
constexpr int kB  = 8;
constexpr int kL  = 512;
constexpr int kD  = 768;
constexpr int kH  = 12;
constexpr int kDK = 64;   // kD / kH

// GEMM tiling: workgroup = 256 threads = 8 waves, wave grid 4(m) x 2(n);
// each wave computes 32x64 (2 A frags x 4 B frags = 8 WMMAs / K-step)
constexpr int BM = 128;
constexpr int BN = 128;
constexpr int BK = 32;
constexpr int LDA_S  = BK + 8;   // A tile stride (f16): 40 -> 80B rows, 16B-aligned chunks
constexpr int LDBT_S = BK + 8;   // transposed-B tile stride (f16), BN rows

typedef __attribute__((ext_vector_type(16))) _Float16 v16h;
typedef __attribute__((ext_vector_type(8)))  _Float16 v8h;
typedef __attribute__((ext_vector_type(8)))  float    v8f;

// TDM descriptor vector types (per compile-probe signatures)
typedef __attribute__((ext_vector_type(4))) unsigned int tdm_v4u;
typedef __attribute__((ext_vector_type(8))) int          tdm_v8i;
typedef __attribute__((ext_vector_type(4))) int          tdm_v4i;

// ---------------------------------------------------------------------------
// WMMA helpers (CDNA5 v_wmma_f32_16x16x32_f16, wave32)
// ---------------------------------------------------------------------------
__device__ __forceinline__ v8f wmma_f16(v16h a, v16h b, v8f c) {
  return __builtin_amdgcn_wmma_f32_16x16x32_f16(
      false, a, false, b, (short)0, c, false, false);
}

// A fragment: lane reads two contiguous 8xf16 (16B) runs of its row -> 2x b128
__device__ __forceinline__ v16h lds_a_frag(const _Float16* As, int mrow) {
  const int lane = threadIdx.x & 31;
  const _Float16* p =
      As + (size_t)(mrow + (lane & 15)) * LDA_S + 8 * (lane >> 4);
  const v8h c0 = *(const v8h*)p;
  const v8h c1 = *(const v8h*)(p + 16);
  return __builtin_shufflevector(c0, c1, 0, 1, 2, 3, 4, 5, 6, 7, 8, 9, 10, 11,
                                 12, 13, 14, 15);
}

// B fragment from transposed LDS tile BsT[n][k]: two 8xf16 (16B) runs -> 2x b128
__device__ __forceinline__ v16h lds_b_frag(const _Float16* BsT, int ncol) {
  const int lane = threadIdx.x & 31;
  const _Float16* p =
      BsT + (size_t)(ncol + (lane & 15)) * LDBT_S + 8 * (lane >> 4);
  const v8h c0 = *(const v8h*)p;
  const v8h c1 = *(const v8h*)(p + 16);
  return __builtin_shufflevector(c0, c1, 0, 1, 2, 3, 4, 5, 6, 7, 8, 9, 10, 11,
                                 12, 13, 14, 15);
}

// 8 WMMAs per wave per K-step: B fragments loaded once, reused for 2 A frags
__device__ __forceinline__ void compute_tile(const _Float16* As,
                                             const _Float16* BsT, int wm,
                                             int wn, v8f acc[2][4]) {
  v16h bf[4];
#pragma unroll
  for (int nt = 0; nt < 4; ++nt) bf[nt] = lds_b_frag(BsT, wn + nt * 16);
#pragma unroll
  for (int mi = 0; mi < 2; ++mi) {
    const v16h af = lds_a_frag(As, wm + mi * 16);
#pragma unroll
    for (int nt = 0; nt < 4; ++nt)
      acc[mi][nt] = wmma_f16(af, bf[nt], acc[mi][nt]);
  }
}

// ---------------------------------------------------------------------------
// staging helpers (fp32 global -> f16 LDS)
// ---------------------------------------------------------------------------
// A tile 128x32 from row-major source (coalesced float4 reads, b64 LDS stores)
__device__ __forceinline__ void stage_a_rm(_Float16* As,
                                           const float* __restrict__ src,
                                           int ld, int tid) {
#pragma unroll
  for (int it = 0; it < 4; ++it) {
    const int row = it * 32 + (tid >> 3);
    const int col = (tid & 7) * 4;
    const float4 v = *(const float4*)&src[(size_t)row * ld + col];
    _Float16* dst = &As[(size_t)row * LDA_S + col];
    dst[0] = (_Float16)v.x; dst[1] = (_Float16)v.y;
    dst[2] = (_Float16)v.z; dst[3] = (_Float16)v.w;
  }
}

// B tile 32(k) x 128(n) from row-major [k][n] source; 4x4 register-block
// transpose: 4 coalesced float4 reads -> 4 contiguous b64 LDS stores
__device__ __forceinline__ void stage_bt_rm(_Float16* BsT,
                                            const float* __restrict__ src,
                                            int ld, int tid) {
  const int k = (tid >> 5) * 4;   // 0..28
  const int n = (tid & 31) * 4;   // 0..124
  float4 r[4];
#pragma unroll
  for (int j = 0; j < 4; ++j)
    r[j] = *(const float4*)&src[(size_t)(k + j) * ld + n];
#pragma unroll
  for (int i = 0; i < 4; ++i) {
    _Float16* dst = &BsT[(size_t)(n + i) * LDBT_S + k];
    dst[0] = (_Float16)(&r[0].x)[i];
    dst[1] = (_Float16)(&r[1].x)[i];
    dst[2] = (_Float16)(&r[2].x)[i];
    dst[3] = (_Float16)(&r[3].x)[i];
  }
}

// B tile when the source is already [n][k] row-major (key^T): straight copy
__device__ __forceinline__ void stage_bt_cm(_Float16* BsT,
                                            const float* __restrict__ src,
                                            int ld, int tid) {
#pragma unroll
  for (int it = 0; it < 4; ++it) {
    const int n = it * 32 + (tid >> 3);
    const int k = (tid & 7) * 4;
    const float4 v = *(const float4*)&src[(size_t)n * ld + k];
    _Float16* dst = &BsT[(size_t)n * LDBT_S + k];
    dst[0] = (_Float16)v.x; dst[1] = (_Float16)v.y;
    dst[2] = (_Float16)v.z; dst[3] = (_Float16)v.w;   // ds_store_b64
  }
}

// ---------------------------------------------------------------------------
// 1) Projection GEMM: out = (x @ W + bias) * scale   [4096 x 768] fp32
// ---------------------------------------------------------------------------
__global__ __launch_bounds__(256)
void proj_gemm(const float* __restrict__ x, const float* __restrict__ W,
               const float* __restrict__ bias, float* __restrict__ out,
               float scale) {
  __shared__ _Float16 As[2][BM * LDA_S];
  __shared__ _Float16 Bs[2][BN * LDBT_S];
  const int n0  = blockIdx.x * BN;
  const int m0  = blockIdx.y * BM;
  const int tid = threadIdx.x;
  const int wid = tid >> 5;
  const int wm  = (wid >> 1) * 32;
  const int wn  = (wid & 1) * 64;

  v8f acc[2][4] = {};
  stage_a_rm(As[0], &x[(size_t)m0 * kD], kD, tid);
  stage_bt_rm(Bs[0], &W[n0], kD, tid);
  const int nsteps = kD / BK;
  for (int s = 0; s < nsteps; ++s) {
    __syncthreads();
    if (s + 1 < nsteps) {
      const int k0 = (s + 1) * BK;
      stage_a_rm(As[(s + 1) & 1], &x[(size_t)m0 * kD + k0], kD, tid);
      stage_bt_rm(Bs[(s + 1) & 1], &W[(size_t)k0 * kD + n0], kD, tid);
      if (s + 2 < nsteps) {  // global_prefetch_b8 for the K tile after next
        __builtin_prefetch(&x[(size_t)(m0 + (tid >> 3)) * kD + k0 + BK], 0, 1);
        __builtin_prefetch(&W[(size_t)(k0 + BK + (tid >> 5)) * kD + n0], 0, 1);
      }
    }
    compute_tile(As[s & 1], Bs[s & 1], wm, wn, acc);
  }
  const int lane = tid & 31;
  const int n = lane & 15, mh = (lane >> 4) * 8;
#pragma unroll
  for (int mi = 0; mi < 2; ++mi) {
#pragma unroll
    for (int nt = 0; nt < 4; ++nt) {
      const int col = n0 + wn + nt * 16 + n;
      const float bv = bias[col];
#pragma unroll
      for (int j = 0; j < 8; ++j)
        out[(size_t)(m0 + wm + mi * 16 + mh + j) * kD + col] =
            (acc[mi][nt][j] + bv) * scale;
    }
  }
}

// ---------------------------------------------------------------------------
// 2) root: R[b,l] = exp(max(x@Wr + br - (mask/-1e4)*50, -40))
// ---------------------------------------------------------------------------
__global__ void root_kernel(const float* __restrict__ x,
                            const float* __restrict__ Wr,
                            const float* __restrict__ br,
                            const float* __restrict__ mask,
                            float* __restrict__ R) {
  const int i = blockIdx.x * blockDim.x + threadIdx.x;
  if (i >= kB * kL) return;
  const float* xr = x + (size_t)i * kD;
  float acc = 0.f;
  for (int d = 0; d < kD; ++d) acc = fmaf(xr[d], Wr[d], acc);
  acc += br[0];
  const float m = mask[i] * (-1.0f / 10000.0f);
  acc = fmaxf(acc - m * 50.0f, -40.0f);
  R[i] = __expf(acc);
}

// ---------------------------------------------------------------------------
// 3) scores -> A = exp(clip(q k^T - 50*(mq+mk), -40))  per (b,h)
// ---------------------------------------------------------------------------
__global__ __launch_bounds__(256)
void scores_kernel(const float* __restrict__ query,
                   const float* __restrict__ key,
                   const float* __restrict__ mask,
                   float* __restrict__ A) {
  __shared__ _Float16 As[2][BM * LDA_S];
  __shared__ _Float16 Bs[2][BN * LDBT_S];
  const int n0 = blockIdx.x * BN;   // key-index tile
  const int q0 = blockIdx.y * BM;   // query tile
  const int bh = blockIdx.z;
  const int b = bh / kH, h = bh % kH;
  const int hoff = h * kDK;
  const int tid = threadIdx.x;
  const int wid = tid >> 5;
  const int wm = (wid >> 1) * 32;
  const int wn = (wid & 1) * 64;
  const float* qsrc = &query[((size_t)b * kL + q0) * kD + hoff];
  const float* ksrc = &key[((size_t)b * kL + n0) * kD + hoff];

  v8f acc[2][4] = {};
  stage_a_rm(As[0], qsrc, kD, tid);
  stage_bt_cm(Bs[0], ksrc, kD, tid);
  const int nsteps = kDK / BK;   // 2
  for (int s = 0; s < nsteps; ++s) {
    __syncthreads();
    if (s + 1 < nsteps) {
      stage_a_rm(As[(s + 1) & 1], qsrc + (s + 1) * BK, kD, tid);
      stage_bt_cm(Bs[(s + 1) & 1], ksrc + (s + 1) * BK, kD, tid);
    }
    compute_tile(As[s & 1], Bs[s & 1], wm, wn, acc);
  }
  const int lane = tid & 31;
  const int n = lane & 15, mh = (lane >> 4) * 8;
  const size_t base = (size_t)bh * kL * kL;
#pragma unroll
  for (int mi = 0; mi < 2; ++mi) {
#pragma unroll
    for (int nt = 0; nt < 4; ++nt) {
      const int kc = n0 + wn + nt * 16 + n;
      const float mk = mask[b * kL + kc] * (-1.0f / 10000.0f);
#pragma unroll
      for (int j = 0; j < 8; ++j) {
        const int qi = q0 + wm + mi * 16 + mh + j;
        const float mq = mask[b * kL + qi] * (-1.0f / 10000.0f);
        const float s2 = fmaxf(acc[mi][nt][j] - mq * 50.f - mk * 50.f, -40.f);
        A[base + (size_t)qi * kL + kc] = __expf(s2);
      }
    }
  }
}

// ---------------------------------------------------------------------------
// 4) colsum[b,h,k] = sum_q A[b,h,q,k]
// ---------------------------------------------------------------------------
__global__ void colsum_kernel(const float* __restrict__ A,
                              float* __restrict__ colsum) {
  const int i = blockIdx.x * blockDim.x + threadIdx.x;
  if (i >= kB * kH * kL) return;
  const int bh = i / kL, kc = i % kL;
  const float* Ab = A + (size_t)bh * kL * kL + kc;
  float s = 0.f;
  for (int q = 0; q < kL; ++q) s += Ab[(size_t)q * kL];
  colsum[i] = s;
}

// ---------------------------------------------------------------------------
// 5) LL = diag(colsum + R) - A
// ---------------------------------------------------------------------------
__global__ void build_ll(const float* __restrict__ A,
                         const float* __restrict__ colsum,
                         const float* __restrict__ R,
                         float* __restrict__ M) {
  const size_t i = (size_t)blockIdx.x * blockDim.x + threadIdx.x;
  if (i >= (size_t)kB * kH * kL * kL) return;
  const int kc = (int)(i % kL);
  const int q  = (int)((i / kL) % kL);
  const int bh = (int)(i / ((size_t)kL * kL));
  const int b  = bh / kH;
  float v = -A[i];
  if (q == kc) v += colsum[bh * kL + kc] + R[b * kL + kc];
  M[i] = v;
}

// ---------------------------------------------------------------------------
// TDM: 1-D tensor load (nelem fp32, contiguous) from global to LDS.
// D# per CDNA5 ISA 8.3/8.4. Tracked by TENSORcnt.
// ---------------------------------------------------------------------------
__device__ __forceinline__ void tdm_load_row(unsigned lds_off,
                                             const float* gptr, int nelem) {
  const unsigned long long ga = (unsigned long long)(size_t)gptr;
  tdm_v4u g0 = {0u, 0u, 0u, 0u};
  g0[0] = 1u;                                     // count=1 valid descriptor
  g0[1] = lds_off;                                // lds_addr (bytes)
  g0[2] = (unsigned)(ga & 0xffffffffu);           // global_addr[31:0]
  g0[3] = (unsigned)((ga >> 32) & 0x01ffffffu)    // global_addr[56:32]
          | (2u << 30);                           // type = 2 ("image")
  tdm_v8i g1 = {0, 0, 0, 0, 0, 0, 0, 0};
  g1[0] = (int)(2u << 16);                        // data_size = 4 bytes
  g1[1] = (int)((unsigned)nelem << 16);           // tensor_dim0[15:0]
  g1[2] = (int)((unsigned)nelem >> 16)            // tensor_dim0[31:16]
          | (int)(1u << 16);                      // tensor_dim1 = 1
  g1[3] = (int)((unsigned)nelem << 16);           // tile_dim0
  g1[5] = nelem;                                  // tensor_dim0_stride
  tdm_v4i z = {0, 0, 0, 0};
#if __clang_major__ >= 23
  tdm_v8i z8 = {0, 0, 0, 0, 0, 0, 0, 0};
  __builtin_amdgcn_tensor_load_to_lds(g0, g1, z, z, z8, 0);
#else
  __builtin_amdgcn_tensor_load_to_lds(g0, g1, z, z, 0);
#endif
}

// ---------------------------------------------------------------------------
// 6) In-place Gauss-Jordan inversion of 96 diagonally-dominant 512x512
//    matrices; 96 MB stays L2-resident (192 MB L2). Pivot row fetched by the
//    Tensor Data Mover (TENSOR_LOAD_TO_LDS + s_wait_tensorcnt).
// ---------------------------------------------------------------------------
__global__ __launch_bounds__(1024)
void gauss_jordan(float* __restrict__ LLbuf) {
  float* M = LLbuf + (size_t)blockIdx.x * kL * kL;
  __shared__ __align__(16) float pivraw[kL];
  __shared__ __align__(16) float pivrow[kL];
  __shared__ __align__(16) float pivcol[kL];
  const int tid = threadIdx.x;
  const int r   = tid >> 1;
  const int ch  = tid & 1;
  const int c0  = ch * (kL / 2);
  // LDS byte offset of pivraw (flat-address low bits = LDS offset)
  const unsigned pivraw_off = (unsigned)(unsigned long long)(void*)&pivraw[0];
  for (int i = 0; i < kL; ++i) {
    if (tid == 0) tdm_load_row(pivraw_off, M + (size_t)i * kL, kL);
    if (tid < kL) pivcol[tid] = M[(size_t)tid * kL + i];
    __builtin_amdgcn_s_wait_tensorcnt(0);
    __syncthreads();
    const float piv = 1.0f / pivcol[i];
    if (tid < kL) pivrow[tid] = pivraw[tid] * piv;
    __syncthreads();
    float4*       Mr = (float4*)(M + (size_t)r * kL + c0);
    const float4* Pr = (const float4*)(pivrow + c0);
    if (r == i) {
#pragma unroll 4
      for (int c = 0; c < kL / 8; ++c) Mr[c] = Pr[c];
      if (i >= c0 && i < c0 + kL / 2) M[(size_t)i * kL + i] = piv;
    } else {
      const float f = pivcol[r];
#pragma unroll 4
      for (int c = 0; c < kL / 8; ++c) {
        float4 m = Mr[c];
        const float4 p = Pr[c];
        m.x -= f * p.x; m.y -= f * p.y; m.z -= f * p.z; m.w -= f * p.w;
        Mr[c] = m;
      }
      if (i >= c0 && i < c0 + kL / 2) M[(size_t)r * kL + i] = -f * piv;
    }
    __syncthreads();
  }
}

// ---------------------------------------------------------------------------
// 7) d0[b,h,l] = R[b,l] * LLinv[b,h,l,l]
// ---------------------------------------------------------------------------
__global__ void d0_kernel(const float* __restrict__ LLinv,
                          const float* __restrict__ R,
                          float* __restrict__ d0) {
  const int i = blockIdx.x * blockDim.x + threadIdx.x;
  if (i >= kB * kH * kL) return;
  const int l = i % kL, bh = i / kL, b = bh / kH;
  d0[i] = R[b * kL + l] * LLinv[(size_t)bh * kL * kL + (size_t)l * kL + l];
}

// ---------------------------------------------------------------------------
// 8) d = A*diag(LLinv)[k] - A*LLinv^T   (in place over the A buffer)
// ---------------------------------------------------------------------------
__global__ void d_kernel(float* __restrict__ dA,
                         const float* __restrict__ LLinv) {
  const size_t i = (size_t)blockIdx.x * blockDim.x + threadIdx.x;
  if (i >= (size_t)kB * kH * kL * kL) return;
  const int kc = (int)(i % kL);
  const int q  = (int)((i / kL) % kL);
  const int bh = (int)(i / ((size_t)kL * kL));
  const float* inv = LLinv + (size_t)bh * kL * kL;
  const float a    = dA[i];
  const float diag = inv[(size_t)kc * kL + kc];
  const float invT = inv[(size_t)kc * kL + q];
  dA[i] = a * diag - a * invT;
}

// ---------------------------------------------------------------------------
// 9) loss_root: masked-mean BCE over heads (single-block reduction)
// ---------------------------------------------------------------------------
__global__ __launch_bounds__(1024)
void loss_kernel(const float* __restrict__ d0,
                 const int* __restrict__ roots_label,
                 float* __restrict__ loss) {
  __shared__ float red[1024];
  float acc = 0.f;
  for (int i = threadIdx.x; i < kB * kL; i += 1024) {
    const float label = (float)roots_label[i];
    const int b = i / kL, l = i % kL;
    float s = 0.f;
#pragma unroll
    for (int h = 0; h < kH; ++h) {
      float p = d0[((size_t)(b * kH + h)) * kL + l];
      p = fminf(fmaxf(p, 1e-5f), 1.f - 1e-5f);
      s += -(label * __logf(p) + (1.f - label) * __logf(1.f - p));
    }
    acc += s * label;
  }
  red[threadIdx.x] = acc;
  __syncthreads();
  for (int s = 512; s > 0; s >>= 1) {
    if (threadIdx.x < s) red[threadIdx.x] += red[threadIdx.x + s];
    __syncthreads();
  }
  if (threadIdx.x == 0) *loss = red[0] / (float)(kB * kL);
}

// ---------------------------------------------------------------------------
// 10) context = (masked d^T) @ x  per (b,h): 512x512 @ 512x768
// ---------------------------------------------------------------------------
// attn tile 128(q) x 32(k): attn[q,k] = d[k,q]; 4x4 register-block transpose
__device__ __forceinline__ void stage_attn(_Float16* As,
                                           const float* __restrict__ dmat,
                                           const float* __restrict__ maskb,
                                           int q0, int k0, int tid) {
  const int k  = (tid >> 5) * 4;   // 0..28
  const int q4 = (tid & 31) * 4;   // 0..124
  float4 r[4];
#pragma unroll
  for (int j = 0; j < 4; ++j) {
    r[j] = *(const float4*)&dmat[(size_t)(k0 + k + j) * kL + q0 + q4];
    if (maskb[k0 + k + j] != 0.f) {
      r[j].x = 0.f; r[j].y = 0.f; r[j].z = 0.f; r[j].w = 0.f;
    }
  }
#pragma unroll
  for (int i = 0; i < 4; ++i) {
    const bool mq = maskb[q0 + q4 + i] != 0.f;
    _Float16* dst = &As[(size_t)(q4 + i) * LDA_S + k];
#pragma unroll
    for (int j = 0; j < 4; ++j)
      dst[j] = (_Float16)(mq ? 0.f : (&r[j].x)[i]);
  }
}

__global__ __launch_bounds__(256)
void context_gemm(const float* __restrict__ d, const float* __restrict__ x,
                  const float* __restrict__ mask, float* __restrict__ ctx) {
  __shared__ _Float16 As[2][BM * LDA_S];
  __shared__ _Float16 Bs[2][BN * LDBT_S];
  const int n0 = blockIdx.x * BN;
  const int q0 = blockIdx.y * BM;
  const int bh = blockIdx.z;
  const int b  = bh / kH;
  const float* dmat  = d + (size_t)bh * kL * kL;
  const float* xb    = x + (size_t)b * kL * kD;
  const float* maskb = mask + b * kL;
  const int tid = threadIdx.x;
  const int wid = tid >> 5;
  const int wm = (wid >> 1) * 32;
  const int wn = (wid & 1) * 64;

  v8f acc[2][4] = {};
  stage_attn(As[0], dmat, maskb, q0, 0, tid);
  stage_bt_rm(Bs[0], &xb[n0], kD, tid);
  const int nsteps = kL / BK;   // 16
  for (int s = 0; s < nsteps; ++s) {
    __syncthreads();
    if (s + 1 < nsteps) {
      const int k0 = (s + 1) * BK;
      stage_attn(As[(s + 1) & 1], dmat, maskb, q0, k0, tid);
      stage_bt_rm(Bs[(s + 1) & 1], &xb[(size_t)k0 * kD + n0], kD, tid);
      if (s + 2 < nsteps) {
        __builtin_prefetch(&dmat[(size_t)(k0 + BK + (tid >> 3)) * kL + q0], 0, 1);
        __builtin_prefetch(&xb[(size_t)(k0 + BK + (tid >> 5)) * kD + n0], 0, 1);
      }
    }
    compute_tile(As[s & 1], Bs[s & 1], wm, wn, acc);
  }
  const int lane = tid & 31;
  const int n = lane & 15, mh = (lane >> 4) * 8;
#pragma unroll
  for (int mi = 0; mi < 2; ++mi) {
#pragma unroll
    for (int nt = 0; nt < 4; ++nt) {
#pragma unroll
      for (int j = 0; j < 8; ++j)
        ctx[((size_t)bh * kL + q0 + wm + mi * 16 + mh + j) * kD + n0 + wn +
            nt * 16 + n] = acc[mi][nt][j];
    }
  }
}

// ---------------------------------------------------------------------------
// host launcher
// ---------------------------------------------------------------------------
extern "C" void kernel_launch(void* const* d_in, const int* in_sizes, int n_in,
                              void* d_out, int out_size, void* d_ws,
                              size_t ws_size, hipStream_t stream) {
  const float* x    = (const float*)d_in[0];
  const float* mask = (const float*)d_in[1];
  const int* roots_label = (const int*)d_in[2];
  // d_in[3] = root_mask (only checked for not-None in reference; unused)
  const float* Wq = (const float*)d_in[4];
  const float* bq = (const float*)d_in[5];
  const float* Wk = (const float*)d_in[6];
  const float* bk = (const float*)d_in[7];
  const float* Wr = (const float*)d_in[8];
  const float* br = (const float*)d_in[9];

  // output layout: context [B,H,L,D] | d [B,H,L,L] | d0 [B,H,L] | loss [1]
  float* out  = (float*)d_out;
  float* ctx  = out;
  float* dmat = ctx + (size_t)kB * kH * kL * kD;   // also used as A (in place)
  float* d0   = dmat + (size_t)kB * kH * kL * kL;
  float* loss = d0 + (size_t)kB * kH * kL;

  // workspace layout
  float* key    = (float*)d_ws;                       // B*L*D
  float* query  = key    + (size_t)kB * kL * kD;      // B*L*D
  float* R      = query  + (size_t)kB * kL * kD;      // B*L
  float* colsum = R      + (size_t)kB * kL;           // B*H*L
  float* LLw    = colsum + (size_t)kB * kH * kL;      // B*H*L*L (96 MB)

  const float qscale = 1.0f / sqrtf((float)kD);
  const size_t nBHLL = (size_t)kB * kH * kL * kL;

  dim3 gp(kD / BN, (kB * kL) / BM);
  proj_gemm<<<gp, 256, 0, stream>>>(x, Wk, bk, key, 1.0f);
  proj_gemm<<<gp, 256, 0, stream>>>(x, Wq, bq, query, qscale);

  root_kernel<<<(kB * kL + 255) / 256, 256, 0, stream>>>(x, Wr, br, mask, R);

  scores_kernel<<<dim3(kL / BN, kL / BM, kB * kH), 256, 0, stream>>>(
      query, key, mask, dmat);

  colsum_kernel<<<(kB * kH * kL + 255) / 256, 256, 0, stream>>>(dmat, colsum);

  build_ll<<<(unsigned)(nBHLL / 256), 256, 0, stream>>>(dmat, colsum, R, LLw);

  gauss_jordan<<<kB * kH, 1024, 0, stream>>>(LLw);

  d0_kernel<<<(kB * kH * kL + 255) / 256, 256, 0, stream>>>(LLw, R, d0);

  d_kernel<<<(unsigned)(nBHLL / 256), 256, 0, stream>>>(dmat, LLw);

  loss_kernel<<<1, 1024, 0, stream>>>(d0, roots_label, loss);

  context_gemm<<<dim3(kD / BN, kL / BM, kB * kH), 256, 0, stream>>>(
      dmat, x, mask, ctx);
}
